// EMAQuantizer_28741921145204
// MI455X (gfx1250) — compile-verified, matrix operand order checked by reading, not run on hardware
//
#include <hip/hip_runtime.h>

// Problem constants (from the reference): x [32,1024,256] f32, codebook [1024,256] f32.
#define DIM    256
#define KCODES 1024

typedef __attribute__((ext_vector_type(8)))  __bf16 v8bf;
typedef __attribute__((ext_vector_type(16))) __bf16 v16bf;
typedef __attribute__((ext_vector_type(8)))  float  v8f;

// ---------------------------------------------------------------------------
// Prep 1: e2[k] = sum_d codebook[k][d]^2, exact f32.
// (x^2 dropped everywhere: constant per row, cannot change the argmin.)
// ---------------------------------------------------------------------------
__global__ void vq_e2_kernel(const float* __restrict__ cb, float* __restrict__ e2) {
    int k = blockIdx.x * blockDim.x + threadIdx.x;
    if (k < KCODES) {
        const float4* row = (const float4*)(cb + (size_t)k * DIM);
        float s = 0.0f;
        #pragma unroll 8
        for (int i = 0; i < DIM / 4; ++i) {
            float4 v = row[i];
            s += v.x * v.x + v.y * v.y + v.z * v.z + v.w * v.w;
        }
        e2[k] = s;
    }
}

// ---------------------------------------------------------------------------
// Prep 2: split codebook into bf16 hi/lo AND pack into WMMA-B fragment order.
// Packed layout (bf16 elements):
//   idx = ((Tg*8 + S)*4 + c)*256 + l*8 + j
//   Tg = 16-col tile (0..63), S = 32-wide D-step (0..7),
//   c  = chunk: 0 = hi K[8*(l>>4)..+7], 1 = hi K[16+8*(l>>4)..+7], 2/3 = lo same,
//   l  = lane (0..31), j = element (0..7)
//   row = Tg*16 + (l&15), col = S*32 + (c&1)*16 + 8*(l>>4) + j
// => every B-fragment load in the GEMM is global_load_b128 at base + lane*16 (coalesced).
// ---------------------------------------------------------------------------
__global__ void vq_pack_cb_kernel(const float* __restrict__ cb,
                                  __bf16* __restrict__ packed) {
    int i = blockIdx.x * blockDim.x + threadIdx.x;   // over 64*8*4*256 = 524288
    int j  = i & 7;
    int l  = (i >> 3) & 31;
    int c  = (i >> 8) & 3;
    int S  = (i >> 10) & 7;
    int Tg = i >> 13;

    int row = Tg * 16 + (l & 15);
    int col = S * 32 + (c & 1) * 16 + 8 * (l >> 4) + j;

    float v = cb[(size_t)row * DIM + col];
    __bf16 h = (__bf16)v;                            // RNE f32 -> bf16
    packed[i] = (c < 2) ? h : (__bf16)(v - (float)h);
}

// ---------------------------------------------------------------------------
// Main kernel: one block = 16 rows x all 1024 codes, 8 waves.
// Wave w owns columns [w*128, w*128+128) as 8 accumulator tiles of 16x16.
// xe = 3x V_WMMA_F32_16X16X32_BF16 per tile per 32-wide D-step (hh + hl + lh),
// tile-pair interleaved so consecutive WMMAs never share an accumulator.
// ---------------------------------------------------------------------------
#define ROWS_PER_BLOCK 16
#define WAVES          8
#define KT_PER_WAVE    8               // 8 tiles * 16 cols = 128 cols / wave
#define XS_STRIDE      (DIM + 8)       // bf16 elems; 528B row stride (16B-aligned, bank-spread)

// A fragment from LDS (ISA 16-bit 16x32 A layout): two contiguous 16B chunks.
__device__ __forceinline__ v16bf load_frag_lds(const __bf16* __restrict__ row,
                                               int dk, int hi) {
    v8bf c0 = *(const v8bf*)(row + dk + 8 * hi);
    v8bf c1 = *(const v8bf*)(row + dk + 16 + 8 * hi);
    return __builtin_shufflevector(c0, c1,
        0, 1, 2, 3, 4, 5, 6, 7, 8, 9, 10, 11, 12, 13, 14, 15);
}

// B fragment from the packed buffer: two 512B-apart... (chunks are +256 elems).
__device__ __forceinline__ v16bf load_frag_packed(const __bf16* __restrict__ p) {
    v8bf c0 = *(const v8bf*)(p);
    v8bf c1 = *(const v8bf*)(p + 256);
    return __builtin_shufflevector(c0, c1,
        0, 1, 2, 3, 4, 5, 6, 7, 8, 9, 10, 11, 12, 13, 14, 15);
}

__global__ __launch_bounds__(256)
void vq_argmin_kernel(const float* __restrict__ x,
                      const float* __restrict__ cb,
                      const __bf16* __restrict__ packed,
                      const float* __restrict__ e2,
                      float* __restrict__ out_q,     // [N, DIM]
                      float* __restrict__ out_idx) { // [N] (indices as float)
    __shared__ alignas(16) __bf16 xh[ROWS_PER_BLOCK * XS_STRIDE];
    __shared__ alignas(16) __bf16 xl[ROWS_PER_BLOCK * XS_STRIDE];
    __shared__ float redDist[WAVES * ROWS_PER_BLOCK];
    __shared__ int   redIdx[WAVES * ROWS_PER_BLOCK];
    __shared__ int   bestRowIdx[ROWS_PER_BLOCK];

    const int tid   = threadIdx.x;
    const int wave  = tid >> 5;
    const int lane  = tid & 31;
    const int lhalf = lane & 15;     // 0..15
    const int hi    = lane >> 4;     // 0/1: K-phase select within fragment
    const int rowBase = blockIdx.x * ROWS_PER_BLOCK;

    // ---- stage x tile into LDS, split into bf16 hi/lo planes (coalesced) ----
    for (int i = tid; i < ROWS_PER_BLOCK * DIM; i += 256) {
        int r = i >> 8;
        int c = i & (DIM - 1);
        float v = x[(size_t)(rowBase + r) * DIM + c];
        __bf16 h = (__bf16)v;
        xh[r * XS_STRIDE + c] = h;
        xl[r * XS_STRIDE + c] = (__bf16)(v - (float)h);
    }
    __syncthreads();

    // ---- WMMA main loop ----
    const int colBase = wave * (KT_PER_WAVE * 16);
    v8f acc[KT_PER_WAVE] = {};

    const __bf16* xhrow = xh + lhalf * XS_STRIDE;
    const __bf16* xlrow = xl + lhalf * XS_STRIDE;
    // per-wave, per-lane base into the packed B buffer (bf16 elements):
    // offset(t,S,c) = wave*65536 + t*8192 + S*1024 + c*256 + lane*8
    const __bf16* bbase = packed + (size_t)wave * 65536 + (size_t)lane * 8;

    for (int S = 0; S < 8; ++S) {
        const int dk = S * 32;
        v16bf ah = load_frag_lds(xhrow, dk, hi);
        v16bf al = load_frag_lds(xlrow, dk, hi);
        const __bf16* bs = bbase + S * 1024;
        #pragma unroll
        for (int t = 0; t < KT_PER_WAVE; t += 2) {
            const __bf16* p0 = bs + t * 8192;
            const __bf16* p1 = bs + (t + 1) * 8192;
            v16bf bh0 = load_frag_packed(p0);
            v16bf bl0 = load_frag_packed(p0 + 512);
            v16bf bh1 = load_frag_packed(p1);
            v16bf bl1 = load_frag_packed(p1 + 512);
            // interleaved: consecutive WMMAs target different accumulators
            acc[t]     = __builtin_amdgcn_wmma_f32_16x16x32_bf16(
                false, ah, false, bh0, (short)0, acc[t],     false, false);
            acc[t + 1] = __builtin_amdgcn_wmma_f32_16x16x32_bf16(
                false, ah, false, bh1, (short)0, acc[t + 1], false, false);
            acc[t]     = __builtin_amdgcn_wmma_f32_16x16x32_bf16(
                false, ah, false, bl0, (short)0, acc[t],     false, false);
            acc[t + 1] = __builtin_amdgcn_wmma_f32_16x16x32_bf16(
                false, ah, false, bl1, (short)0, acc[t + 1], false, false);
            acc[t]     = __builtin_amdgcn_wmma_f32_16x16x32_bf16(
                false, al, false, bh0, (short)0, acc[t],     false, false);
            acc[t + 1] = __builtin_amdgcn_wmma_f32_16x16x32_bf16(
                false, al, false, bh1, (short)0, acc[t + 1], false, false);
        }
    }

    // ---- per-thread argmin over this wave's 128 columns ----
    // C/D layout: VGPR j -> row (j + 8*hi), col = colBase + t*16 + lhalf.
    float bestD[8];
    int   bestI[8];
    #pragma unroll
    for (int j = 0; j < 8; ++j) { bestD[j] = 3.4e38f; bestI[j] = 0x7FFFFFFF; }

    #pragma unroll
    for (int t = 0; t < KT_PER_WAVE; ++t) {
        int col = colBase + t * 16 + lhalf;
        float e2v = e2[col];
        #pragma unroll
        for (int j = 0; j < 8; ++j) {
            float d = e2v - 2.0f * acc[t][j];
            if (d < bestD[j]) { bestD[j] = d; bestI[j] = col; }
        }
    }

    // ---- xor-shuffle reduction within each 16-lane half (wave32) ----
    #pragma unroll
    for (int off = 1; off < 16; off <<= 1) {
        #pragma unroll
        for (int j = 0; j < 8; ++j) {
            float od = __shfl_xor(bestD[j], off, 32);
            int   oi = __shfl_xor(bestI[j], off, 32);
            if (od < bestD[j] || (od == bestD[j] && oi < bestI[j])) {
                bestD[j] = od; bestI[j] = oi;
            }
        }
    }

    // lane 0 holds rows 0..7, lane 16 holds rows 8..15
    if (lhalf == 0) {
        #pragma unroll
        for (int j = 0; j < 8; ++j) {
            int row = j + 8 * hi;
            redDist[wave * ROWS_PER_BLOCK + row] = bestD[j];
            redIdx [wave * ROWS_PER_BLOCK + row] = bestI[j];
        }
    }
    __syncthreads();

    // ---- cross-wave reduction: one thread per row ----
    if (tid < ROWS_PER_BLOCK) {
        float bd = redDist[tid];
        int   bi = redIdx[tid];
        #pragma unroll
        for (int w = 1; w < WAVES; ++w) {
            float d  = redDist[w * ROWS_PER_BLOCK + tid];
            int   i2 = redIdx [w * ROWS_PER_BLOCK + tid];
            if (d < bd || (d == bd && i2 < bi)) { bd = d; bi = i2; }
        }
        bestRowIdx[tid] = bi;
        out_idx[rowBase + tid] = (float)bi;
    }
    __syncthreads();

    // ---- gather winning codebook rows (original f32!) -> x_q output ----
    for (int i = tid; i < ROWS_PER_BLOCK * DIM; i += 256) {
        int r = i >> 8;
        int c = i & (DIM - 1);
        out_q[(size_t)(rowBase + r) * DIM + c] =
            cb[(size_t)bestRowIdx[r] * DIM + c];
    }
}

// ---------------------------------------------------------------------------
extern "C" void kernel_launch(void* const* d_in, const int* in_sizes, int n_in,
                              void* d_out, int out_size, void* d_ws, size_t ws_size,
                              hipStream_t stream) {
    const float* x  = (const float*)d_in[0];   // [32,1024,256] f32
    const float* cb = (const float*)d_in[1];   // [1024,256]   f32

    const int nrows = in_sizes[0] / DIM;       // 32768

    // Workspace layout: e2 f32[1024] (4KB pad) | packed bf16[524288] (1MB)
    float*  e2     = (float*)d_ws;
    __bf16* packed = (__bf16*)((char*)d_ws + 4096);

    float* out_q   = (float*)d_out;                  // nrows*DIM floats
    float* out_idx = out_q + (size_t)nrows * DIM;    // then nrows index floats

    vq_e2_kernel<<<(KCODES + 255) / 256, 256, 0, stream>>>(cb, e2);
    vq_pack_cb_kernel<<<(64 * 8 * 4 * 256) / 256, 256, 0, stream>>>(cb, packed);
    vq_argmin_kernel<<<nrows / ROWS_PER_BLOCK, 256, 0, stream>>>(
        x, cb, packed, e2, out_q, out_idx);
}
